// DeepEGNN_44796508897964
// MI455X (gfx1250) — matrix operand order repaired and checked by reading, not compiled, not run on hardware
//
#include <hip/hip_runtime.h>
#include <math.h>

// ---------------- problem constants ----------------
#define NN 50000
#define EE 800000
#define FF 64      // node feature dim
#define MM 64      // message dim
#define HH 128     // hidden dim
#define DD 3       // coordinate dim
#define EIN 136    // 2*F + NUM_ENCODE
#define KP 160     // EIN padded up to multiple of 32 for WMMA K-steps
#define LL 4       // layers

// fragment-image sizes (in halves); 512 halves per 16x16x32 B-fragment tile
#define EW_TILES 76        // 5*8 (We1) + 4*4 (We2) + 2*8 (Wp1) + 4 (Wp2 col0)
#define NW_TILES 48        // 4*8 (Wn1) + 4*4 (Wn2)
#define EW_HALVES (EW_TILES * 512)
#define NW_HALVES (NW_TILES * 512)

typedef _Float16 v16h __attribute__((ext_vector_type(16)));
typedef _Float16 v8h  __attribute__((ext_vector_type(8)));
typedef _Float16 v4h  __attribute__((ext_vector_type(4)));
typedef float    v8f  __attribute__((ext_vector_type(8)));
typedef unsigned int v4u __attribute__((ext_vector_type(4)));
typedef int      v8i  __attribute__((ext_vector_type(8)));
typedef int      v4i  __attribute__((ext_vector_type(4)));

// Branch-free tanh-GELU:  0.5 v (1+tanh(c(v+0.044715 v^3))) == v * sigmoid(2c(...))
__device__ __forceinline__ float gelu_tanh(float v) {
    float u = 1.5957691216057308f * v * __builtin_fmaf(0.044715f, v * v, 1.0f);
    float e = __builtin_amdgcn_exp2f(-1.4426950408889634f * u);   // v_exp_f32
    return v * __builtin_amdgcn_rcpf(1.0f + e);                   // v_rcp_f32
}

// ---- WMMA fragment loaders (layouts per CDNA5 ISA §7.12.2) ----
__device__ __forceinline__ v16h load_a_frag(const _Float16* At, int ldk, int k0, int lane) {
    const int row = lane & 15, kh = lane >> 4;
    const _Float16* base = At + row * ldk + k0 + (kh << 3);
    v8h lo = *(const v8h*)base;            // ds_load_b128
    v8h hi = *(const v8h*)(base + 16);     // ds_load_b128
    return __builtin_shufflevector(lo, hi, 0, 1, 2, 3, 4, 5, 6, 7,
                                           8, 9, 10, 11, 12, 13, 14, 15);
}

__device__ __forceinline__ v16h load_b_fragf(const _Float16* frag, int tileIdx, int lane) {
    return *(const v16h*)(frag + (tileIdx << 9) + (lane << 4));
}

#define WMMA_F16(A, B, C) \
    __builtin_amdgcn_wmma_f32_16x16x32_f16(false, (A), false, (B), (short)0, (C), false, false)

// ---- Tensor Data Mover: 1-D bulk copy global -> LDS (D# per ISA §8.3/8.4) ----
// bytes must be a multiple of 8; issued once per wave (TDM ignores EXEC).
// This toolchain exposes the 6-arg builtin: (v4u, v8i, v4i, v4i, v8i, i32 cpol).
__device__ __forceinline__ void tdm_load_1d(const void* gsrc, void* lds_dst, unsigned bytes) {
    unsigned long long ga = (unsigned long long)gsrc;
    unsigned lds = (unsigned)(unsigned long long)lds_dst;  // LDS aperture: offset in [31:0]
    unsigned nel = bytes >> 3;                             // 8-byte elements
    v4u g0;
    g0[0] = 1u;                                            // count=1 (valid user D#)
    g0[1] = lds;                                           // lds_addr
    g0[2] = (unsigned)(ga & 0xffffffffu);                  // global_addr[31:0]
    g0[3] = (unsigned)((ga >> 32) & 0x01ffffffu) | (2u << 30); // global_addr[56:32] | type=2
    v8i g1;
    g1[0] = (int)(3u << 16);                               // data_size=3 (8B)
    g1[1] = (int)((nel & 0xffffu) << 16);                  // tensor_dim0[15:0]  (bits 63:48)
    g1[2] = (int)((nel >> 16) & 0xffffu);                  // tensor_dim0[31:16]
    g1[3] = (int)((nel & 0xffffu) << 16);                  // tile_dim0 (bits 127:112)
    g1[4] = 0;                                             // tile_dim1/2 unused
    g1[5] = (int)nel;                                      // tensor_dim0_stride[31:0]
    g1[6] = 0;
    g1[7] = 0;
    v4i g2 = {0, 0, 0, 0};
    v4i g3 = {0, 0, 0, 0};
    v8i g4 = {0, 0, 0, 0, 0, 0, 0, 0};
    __builtin_amdgcn_tensor_load_to_lds(g0, g1, g2, g3, g4, 0);
}

// =====================================================================
// Prep kernels: build f16 fragment-major weight images in global memory
//   frag[(kt*NT + nt)*512 + l*16 + e] = W[(kt*32 + (e>=8?16:0) + (l>=16?8:0) + (e&7))*ldn + nt*16 + (l&15)]
// =====================================================================
__global__ void prep_frag_kernel(const float* __restrict__ W, _Float16* __restrict__ dst,
                                 int KT, int NT, int ldn, int Kreal) {
    int i = blockIdx.x * blockDim.x + threadIdx.x;
    if (i >= KT * NT * 512) return;
    int t = i >> 9, w = i & 511, l = w >> 4, e2 = w & 15;
    int kt = t / NT, nt = t - kt * NT;
    int K = kt * 32 + ((e2 >> 3) << 4) + ((l >> 4) << 3) + (e2 & 7);
    int n = (nt << 4) + (l & 15);
    dst[i] = (K < Kreal) ? (_Float16)W[(size_t)K * ldn + n] : (_Float16)0.0f;
}

__global__ void prep_vec_kernel(const float* __restrict__ wvec, _Float16* __restrict__ dst,
                                int KT) {  // column-0-only B matrix from a K-vector
    int i = blockIdx.x * blockDim.x + threadIdx.x;
    if (i >= KT * 512) return;
    int t = i >> 9, w = i & 511, l = w >> 4, e2 = w & 15;
    int K = t * 32 + ((e2 >> 3) << 4) + ((l >> 4) << 3) + (e2 & 7);
    dst[i] = ((l & 15) == 0) ? (_Float16)wvec[K] : (_Float16)0.0f;
}

// =====================================================================
// Edge kernel: per 16-edge tile, fused  edge-MLP -> m -> (agg atomics)
//                                       -> p-MLP -> wgt -> (posacc atomics)
// 4 waves / block; weights arrive via one TDM DMA into LDS.
// =====================================================================
__global__ __launch_bounds__(128, 1) void egnn_edge_kernel(
    const float* __restrict__ x, const float* __restrict__ pos,
    const int* __restrict__ srcI, const int* __restrict__ dstI,
    const _Float16* __restrict__ wfrag,
    const float* __restrict__ be1, const float* __restrict__ be2,
    const float* __restrict__ bp1, const float* __restrict__ bp2v,
    float* __restrict__ agg, float* __restrict__ posacc)
{
    __shared__ __align__(32) _Float16 sWf[EW_HALVES];      // 77824 B, one TDM load
    __shared__ float    sBe1[HH], sBe2[MM], sBp1[HH];
    __shared__ float    sBp2;
    __shared__ __align__(32) _Float16 sA [4][16 * KP];     // 20480 B per-wave A tiles
    __shared__ __align__(32) _Float16 sHd[4][16 * HH];     // 16384 B per-wave hidden
    __shared__ __align__(32) _Float16 sMt[4][16 * MM];     //  8192 B per-wave messages
    __shared__ float    sRel[4][16][3];
    __shared__ int      sDst[4][16];
    __shared__ float    sWgt[4][16];

    const int tid = threadIdx.x;

    // ---- wave 0 DMAs the whole pre-swizzled weight image into LDS ----
    if (tid < 32) {
        tdm_load_1d(wfrag, sWf, EW_HALVES * 2);
        __builtin_amdgcn_s_wait_tensorcnt(0);
    }
    for (int i = tid; i < HH; i += 128) { sBe1[i] = be1[i]; sBp1[i] = bp1[i]; }
    for (int i = tid; i < MM; i += 128) sBe2[i] = be2[i];
    if (tid == 0) sBp2 = bp2v[0];
    __syncthreads();

    const _Float16* sWe1f = sWf;                 // 5*8 tiles
    const _Float16* sWe2f = sWf + 40 * 512;      // 4*4 tiles
    const _Float16* sWp1f = sWf + 56 * 512;      // 2*8 tiles
    const _Float16* sWp2f = sWf + 72 * 512;      // 4 tiles (col 0 = wp2)

    const int wave = tid >> 5, lane = tid & 31;
    const int row  = lane & 15, half = lane >> 4, col = lane & 15;
    _Float16* At = sA[wave];
    _Float16* Hd = sHd[wave];
    _Float16* Mt = sMt[wave];

    // hoist the loop-invariant wp2 fragments into registers (4 x v16h)
    v16h wp2f[4];
#pragma unroll
    for (int kt = 0; kt < 4; ++kt) wp2f[kt] = load_b_fragf(sWp2f, kt, lane);

    const int numTiles  = (EE + 15) / 16;                 // 50000
    const int gridWaves = gridDim.x * 4;
    const int wg        = blockIdx.x * 4 + wave;
    const int nIter     = (numTiles + gridWaves - 1) / gridWaves;

    for (int it = 0; it < nIter; ++it) {
        const int tile  = wg + it * gridWaves;
        const bool tval = (tile < numTiles);
        const int  e    = tile * 16 + row;
        const bool eval = tval && (e < EE);
        int d = 0, s = 0;
        if (eval) { s = srcI[e]; d = dstI[e]; }

        // speculative prefetch of next iteration's edge indices (global_prefetch_b8)
        {
            const int enext = (tile + gridWaves) * 16 + row;
            if (enext < EE) {
                __builtin_prefetch(srcI + enext, 0, 1);
                __builtin_prefetch(dstI + enext, 0, 1);
            }
        }

        // ---- stage A tile: [x[dst] | x[src] | fourier(dist) | pad] ----
        if (half == 0) {
            sDst[wave][row] = eval ? d : -1;
            const float4* xd4 = (const float4*)(x + (size_t)d * FF);
#pragma unroll
            for (int c4 = 0; c4 < FF / 4; ++c4) {
                float4 v = xd4[c4];
                v4h h4 = { (_Float16)v.x, (_Float16)v.y, (_Float16)v.z, (_Float16)v.w };
                *(v4h*)&At[row * KP + c4 * 4] = h4;
            }
        } else {
            const float4* xs4 = (const float4*)(x + (size_t)s * FF);
#pragma unroll
            for (int c4 = 0; c4 < FF / 4; ++c4) {
                float4 v = xs4[c4];
                v4h h4 = { (_Float16)v.x, (_Float16)v.y, (_Float16)v.z, (_Float16)v.w };
                *(v4h*)&At[row * KP + FF + c4 * 4] = h4;
            }
            float dx = pos[(size_t)d * 3 + 0] - pos[(size_t)s * 3 + 0];
            float dy = pos[(size_t)d * 3 + 1] - pos[(size_t)s * 3 + 1];
            float dz = pos[(size_t)d * 3 + 2] - pos[(size_t)s * 3 + 2];
            sRel[wave][row][0] = dx; sRel[wave][row][1] = dy; sRel[wave][row][2] = dz;
            float sc = dx * dx + dy * dy + dz * dz;
#pragma unroll
            for (int k = 0; k < 4; ++k) {            // [sin(s0..3), cos(s0..3)]
                At[row * KP + 128 + k] = (_Float16)__sinf(sc);
                At[row * KP + 132 + k] = (_Float16)__cosf(sc);
                sc *= 0.5f;
            }
            v4h z4 = { (_Float16)0, (_Float16)0, (_Float16)0, (_Float16)0 };
#pragma unroll
            for (int c = EIN; c < KP; c += 4) *(v4h*)&At[row * KP + c] = z4;
        }
        __syncthreads();

        // ---- MLP-1: hidden = gelu(A(16x136) @ We1(136x128) + b1) ----
#pragma unroll
        for (int nt = 0; nt < HH / 16; ++nt) {
            v8f acc = {};
#pragma unroll
            for (int kt = 0; kt < KP / 32; ++kt) {
                v16h a = load_a_frag(At, KP, kt * 32, lane);
                v16h b = load_b_fragf(sWe1f, kt * 8 + nt, lane);
                acc = WMMA_F16(a, b, acc);
            }
            float bias = sBe1[nt * 16 + col];
#pragma unroll
            for (int r = 0; r < 8; ++r)
                Hd[(r + half * 8) * HH + nt * 16 + col] = (_Float16)gelu_tanh(acc[r] + bias);
        }
        __syncthreads();

        // ---- MLP-2: m = gelu(H(16x128) @ We2(128x64) + b2); scatter agg ----
#pragma unroll
        for (int nt = 0; nt < MM / 16; ++nt) {
            v8f acc = {};
#pragma unroll
            for (int kt = 0; kt < HH / 32; ++kt) {
                v16h a = load_a_frag(Hd, HH, kt * 32, lane);
                v16h b = load_b_fragf(sWe2f, kt * 4 + nt, lane);
                acc = WMMA_F16(a, b, acc);
            }
            float bias = sBe2[nt * 16 + col];
#pragma unroll
            for (int r = 0; r < 8; ++r) {
                int rr = r + half * 8;
                float mv = gelu_tanh(acc[r] + bias);
                Mt[rr * MM + nt * 16 + col] = (_Float16)mv;
                int dd = sDst[wave][rr];
                if (dd >= 0) atomicAdd(&agg[(size_t)dd * MM + nt * 16 + col], mv);
            }
        }
        __syncthreads();

        // ---- p-MLP hidden: h2 = gelu(m(16x64) @ Wp1(64x128) + bp1) (reuse Hd) ----
#pragma unroll
        for (int nt = 0; nt < HH / 16; ++nt) {
            v8f acc = {};
#pragma unroll
            for (int kt = 0; kt < MM / 32; ++kt) {
                v16h a = load_a_frag(Mt, MM, kt * 32, lane);
                v16h b = load_b_fragf(sWp1f, kt * 8 + nt, lane);
                acc = WMMA_F16(a, b, acc);
            }
            float bias = sBp1[nt * 16 + col];
#pragma unroll
            for (int r = 0; r < 8; ++r)
                Hd[(r + half * 8) * HH + nt * 16 + col] = (_Float16)gelu_tanh(acc[r] + bias);
        }
        __syncthreads();

        // ---- wgt = h2 @ wp2 + bp2 via WMMA against the col-0-only B matrix ----
        {
            v8f accw = {};
#pragma unroll
            for (int kt = 0; kt < 4; ++kt) {
                v16h a = load_a_frag(Hd, HH, kt * 32, lane);
                accw = WMMA_F16(a, wp2f[kt], accw);
            }
            if (col == 0) {
#pragma unroll
                for (int r = 0; r < 8; ++r)
                    sWgt[wave][r + half * 8] = accw[r] + sBp2;
            }
        }
        __syncthreads();

        // ---- scatter rel * wgt into posacc ----
        if (lane < 16) {
            int dd = sDst[wave][lane];
            if (dd >= 0) {
                float w = sWgt[wave][lane];
                float* pa = posacc + (size_t)dd * 3;
                atomicAdd(&pa[0], sRel[wave][lane][0] * w);
                atomicAdd(&pa[1], sRel[wave][lane][1] * w);
                atomicAdd(&pa[2], sRel[wave][lane][2] * w);
            }
        }
        __syncthreads();
    }
}

// =====================================================================
// Node kernel: x_new = gelu([x|agg](16x128) @ Wn1) @ Wn2 + b2
// =====================================================================
__global__ __launch_bounds__(128, 1) void egnn_node_kernel(
    const float* __restrict__ x, const float* __restrict__ agg,
    const _Float16* __restrict__ wfrag,
    const float* __restrict__ bn1, const float* __restrict__ bn2,
    float* __restrict__ xout)
{
    __shared__ __align__(32) _Float16 sWf[NW_HALVES];      // 49152 B, one TDM load
    __shared__ float    sB1[HH], sB2[FF];
    __shared__ __align__(32) _Float16 sA [4][16 * HH];     // 16384 B
    __shared__ __align__(32) _Float16 sHd[4][16 * HH];     // 16384 B

    const int tid = threadIdx.x;
    if (tid < 32) {
        tdm_load_1d(wfrag, sWf, NW_HALVES * 2);
        __builtin_amdgcn_s_wait_tensorcnt(0);
    }
    for (int i = tid; i < HH; i += 128) sB1[i] = bn1[i];
    for (int i = tid; i < FF; i += 128) sB2[i] = bn2[i];
    __syncthreads();

    const _Float16* sW1f = sWf;                 // 4*8 tiles
    const _Float16* sW2f = sWf + 32 * 512;      // 4*4 tiles

    const int wave = tid >> 5, lane = tid & 31;
    const int row  = lane & 15, half = lane >> 4, col = lane & 15;
    _Float16* At = sA[wave];
    _Float16* Hd = sHd[wave];

    const int numTiles  = (NN + 15) / 16;                 // 3125
    const int gridWaves = gridDim.x * 4;
    const int wg        = blockIdx.x * 4 + wave;
    const int nIter     = (numTiles + gridWaves - 1) / gridWaves;

    for (int it = 0; it < nIter; ++it) {
        const int tile  = wg + it * gridWaves;
        const bool tval = (tile < numTiles);
        int node = tile * 16 + row;
        const bool nval = tval && (node < NN);
        if (!nval) node = 0;

        const float4* src4 = (const float4*)((half == 0 ? x : agg) + (size_t)node * FF);
        const int cbase = half == 0 ? 0 : FF;
#pragma unroll
        for (int c4 = 0; c4 < FF / 4; ++c4) {
            float4 v = src4[c4];
            v4h h4 = { (_Float16)v.x, (_Float16)v.y, (_Float16)v.z, (_Float16)v.w };
            *(v4h*)&At[row * HH + cbase + c4 * 4] = h4;
        }
        __syncthreads();

#pragma unroll
        for (int nt = 0; nt < HH / 16; ++nt) {
            v8f acc = {};
#pragma unroll
            for (int kt = 0; kt < HH / 32; ++kt) {
                v16h a = load_a_frag(At, HH, kt * 32, lane);
                v16h b = load_b_fragf(sW1f, kt * 8 + nt, lane);
                acc = WMMA_F16(a, b, acc);
            }
            float bias = sB1[nt * 16 + col];
#pragma unroll
            for (int r = 0; r < 8; ++r)
                Hd[(r + half * 8) * HH + nt * 16 + col] = (_Float16)gelu_tanh(acc[r] + bias);
        }
        __syncthreads();

#pragma unroll
        for (int nt = 0; nt < FF / 16; ++nt) {
            v8f acc = {};
#pragma unroll
            for (int kt = 0; kt < HH / 32; ++kt) {
                v16h a = load_a_frag(Hd, HH, kt * 32, lane);
                v16h b = load_b_fragf(sW2f, kt * 4 + nt, lane);
                acc = WMMA_F16(a, b, acc);
            }
            float bias = sB2[nt * 16 + col];
#pragma unroll
            for (int r = 0; r < 8; ++r) {
                int rr = r + half * 8;
                int nd = tile * 16 + rr;
                if (tval && nd < NN)
                    xout[(size_t)nd * FF + nt * 16 + col] = acc[r] + bias;
            }
        }
        __syncthreads();
    }
}

// ---------------- small helper kernels ----------------
__global__ void egnn_cnt_kernel(const int* __restrict__ dstI, float* __restrict__ cnt) {
    int i = blockIdx.x * blockDim.x + threadIdx.x;
    if (i < EE) atomicAdd(&cnt[dstI[i]], 1.0f);
}

__global__ void egnn_pos_kernel(const float* __restrict__ posin, const float* __restrict__ posacc,
                                const float* __restrict__ cnt, float* __restrict__ posout) {
    int i = blockIdx.x * blockDim.x + threadIdx.x;
    if (i < NN * DD) {
        float c = cnt[i / 3];
        c = c < 1.0f ? 1.0f : c;
        posout[i] = posin[i] + posacc[i] * __builtin_amdgcn_rcpf(c);
    }
}

// =====================================================================
// Host launch
// =====================================================================
extern "C" void kernel_launch(void* const* d_in, const int* in_sizes, int n_in,
                              void* d_out, int out_size, void* d_ws, size_t ws_size,
                              hipStream_t stream) {
    (void)in_sizes; (void)n_in; (void)out_size; (void)ws_size;

    const float* x_in   = (const float*)d_in[0];
    const float* pos_in = (const float*)d_in[1];
    const int*   eidx   = (const int*)  d_in[2];
    const float* w_e1 = (const float*)d_in[3];  const float* b_e1 = (const float*)d_in[4];
    const float* w_e2 = (const float*)d_in[5];  const float* b_e2 = (const float*)d_in[6];
    const float* w_n1 = (const float*)d_in[7];  const float* b_n1 = (const float*)d_in[8];
    const float* w_n2 = (const float*)d_in[9];  const float* b_n2 = (const float*)d_in[10];
    const float* w_p1 = (const float*)d_in[11]; const float* b_p1 = (const float*)d_in[12];
    const float* w_p2 = (const float*)d_in[13]; const float* b_p2 = (const float*)d_in[14];

    const int* srcI = eidx;
    const int* dstI = eidx + EE;

    float* wsF  = (float*)d_ws;
    float* cnt  = wsF;                 wsF += NN;
    float* xa   = wsF;                 wsF += (size_t)NN * FF;
    float* xb   = wsF;                 wsF += (size_t)NN * FF;
    float* pa   = wsF;                 wsF += (size_t)NN * DD;
    float* pb   = wsF;                 wsF += (size_t)NN * DD;
    float* agg  = wsF;                 wsF += (size_t)NN * MM;
    float* pacc = wsF;                 wsF += (size_t)NN * DD;

    // 256B-aligned region for pre-swizzled f16 weight images
    size_t off = (size_t)((char*)wsF - (char*)d_ws);
    off = (off + 255) & ~(size_t)255;
    _Float16* edgeWf = (_Float16*)((char*)d_ws + off);           // LL * EW_HALVES
    _Float16* nodeWf = edgeWf + (size_t)LL * EW_HALVES;          // LL * NW_HALVES

    // ---- build fragment-major weight images (once, tiny) ----
    for (int l = 0; l < LL; ++l) {
        _Float16* ef = edgeWf + (size_t)l * EW_HALVES;
        prep_frag_kernel<<<(40 * 512 + 255) / 256, 256, 0, stream>>>(
            w_e1 + (size_t)l * EIN * HH, ef,            5, 8, HH, EIN);
        prep_frag_kernel<<<(16 * 512 + 255) / 256, 256, 0, stream>>>(
            w_e2 + (size_t)l * HH * MM,  ef + 40 * 512, 4, 4, MM, HH);
        prep_frag_kernel<<<(16 * 512 + 255) / 256, 256, 0, stream>>>(
            w_p1 + (size_t)l * MM * HH,  ef + 56 * 512, 2, 8, HH, MM);
        prep_vec_kernel<<<(4 * 512 + 255) / 256, 256, 0, stream>>>(
            w_p2 + (size_t)l * HH,       ef + 72 * 512, 4);
        _Float16* nf = nodeWf + (size_t)l * NW_HALVES;
        prep_frag_kernel<<<(32 * 512 + 255) / 256, 256, 0, stream>>>(
            w_n1 + (size_t)l * HH * HH,  nf,            4, 8, HH, HH);
        prep_frag_kernel<<<(16 * 512 + 255) / 256, 256, 0, stream>>>(
            w_n2 + (size_t)l * HH * FF,  nf + 32 * 512, 4, 4, FF, HH);
    }

    // degree counts (once)
    (void)hipMemsetAsync(cnt, 0, (size_t)NN * sizeof(float), stream);
    egnn_cnt_kernel<<<(EE + 255) / 256, 256, 0, stream>>>(dstI, cnt);

    // bring state into ping-pong buffers
    (void)hipMemcpyAsync(xa, x_in,  (size_t)NN * FF * sizeof(float), hipMemcpyDeviceToDevice, stream);
    (void)hipMemcpyAsync(pa, pos_in,(size_t)NN * DD * sizeof(float), hipMemcpyDeviceToDevice, stream);

    float* xcur = xa; float* xnxt = xb;
    float* pcur = pa; float* pnxt = pb;

    for (int l = 0; l < LL; ++l) {
        (void)hipMemsetAsync(agg,  0, (size_t)NN * MM * sizeof(float), stream);
        (void)hipMemsetAsync(pacc, 0, (size_t)NN * DD * sizeof(float), stream);

        egnn_edge_kernel<<<1024, 128, 0, stream>>>(
            xcur, pcur, srcI, dstI,
            edgeWf + (size_t)l * EW_HALVES,
            b_e1 + (size_t)l * HH, b_e2 + (size_t)l * MM,
            b_p1 + (size_t)l * HH, b_p2 + (size_t)l,
            agg, pacc);

        egnn_pos_kernel<<<(NN * DD + 255) / 256, 256, 0, stream>>>(pcur, pacc, cnt, pnxt);

        egnn_node_kernel<<<512, 128, 0, stream>>>(
            xcur, agg,
            nodeWf + (size_t)l * NW_HALVES,
            b_n1 + (size_t)l * HH, b_n2 + (size_t)l * FF,
            xnxt);

        float* t;
        t = xcur; xcur = xnxt; xnxt = t;
        t = pcur; pcur = pnxt; pnxt = t;
    }

    (void)hipMemcpyAsync(d_out, xcur, (size_t)NN * FF * sizeof(float), hipMemcpyDeviceToDevice, stream);
    (void)hipMemcpyAsync((float*)d_out + (size_t)NN * FF, pcur,
                         (size_t)NN * DD * sizeof(float), hipMemcpyDeviceToDevice, stream);
}
